// LSTMModel_OnetoMany_74517682585644
// MI455X (gfx1250) — matrix operand order repaired and checked by reading, not compiled, not run on hardware
//
#include <hip/hip_runtime.h>

// ---------------------------------------------------------------------------
// 2-layer LSTM (B=512, H=1024, T=106) for MI455X / gfx1250, wave32.
// bf16 V_WMMA_F32_16X16X32_BF16 GEMMs with fused gate nonlinearity epilogue.
// Round 4: enforce the depth-4 software pipeline with hard
// __builtin_amdgcn_sched_barrier(0) fences. Previous rounds showed the
// scheduler re-pairing each WMMA group with its adjacent loads (wait-0 on L2
// every iteration). With hard fences, slot kc%4 is consumed by 4 WMMAs and
// only then refilled with kc+4's fragments; 4 slot values stay live so the
// allocator keeps 4 distinct register sets and waits run ~24 loads deep.
// ---------------------------------------------------------------------------

typedef __attribute__((ext_vector_type(16))) __bf16 v16bf;
typedef __attribute__((ext_vector_type(8)))  __bf16 v8bf;
typedef __attribute__((ext_vector_type(8)))  float  v8f;
typedef __attribute__((ext_vector_type(4)))  float  v4f;

static constexpr int Bsz = 512;
static constexpr int Hd  = 1024;
static constexpr int FH  = 4096;   // 4*H
static constexpr int T   = 106;

__device__ __forceinline__ float sigf(float x) {
  return 1.0f / (1.0f + __expf(-x));
}
__device__ __forceinline__ float tanhf_(float x) {
  float e = __expf(-2.0f * x);
  return (1.0f - e) / (1.0f + e);
}

// ---------------------------------------------------------------------------
// Prep: zero recurrent state, build b1 = b_ih1 + b_hh1 (gate-major [4][H]).
// ---------------------------------------------------------------------------
__global__ void init_state(float* c0, float* c1, __bf16* h0bf0, __bf16* h1bf0,
                           const float* __restrict__ b_ih1,
                           const float* __restrict__ b_hh1,
                           float* __restrict__ b1p) {
  int i = blockIdx.x * blockDim.x + threadIdx.x;   // 0 .. B*H-1
  c0[i] = 0.0f;
  c1[i] = 0.0f;
  h0bf0[i] = (__bf16)0.0f;
  h1bf0[i] = (__bf16)0.0f;
  if (i < FH) b1p[i] = b_ih1[i] + b_hh1[i];
}

// ---------------------------------------------------------------------------
// Prep: fp32 W[4H][H] -> bf16 WMMA B-fragments.
// frag index = ((hb*4 + g)*32 + kc)*512 + lane*16 + e
//   column n = lane%16 (h = hb*16+n, permuted so gate g owns a 16-col tile)
//   k        = kc*32 + (lane/16)*8 + (e%8) + (e/8)*16
// ---------------------------------------------------------------------------
__global__ void swizzle_w(const float* __restrict__ W, __bf16* __restrict__ frag) {
  int tid = blockIdx.x * blockDim.x + threadIdx.x;  // 0 .. 4M-1
  int f    = tid & 511;
  int kc   = (tid >> 9) & 31;
  int g    = (tid >> 14) & 3;
  int hb   = tid >> 16;                             // 0..63
  int lane = f >> 4;
  int e    = f & 15;
  int n    = lane & 15;
  int k    = kc * 32 + ((lane >> 4) << 3) + (e & 7) + ((e >> 3) << 4);
  int row  = g * Hd + hb * 16 + n;                  // original row of W
  frag[tid] = (__bf16)W[row * Hd + k];
}

// ---------------------------------------------------------------------------
// Prep: xproj = x @ W_ih0^T + b_ih0 + b_hh0, stored in C-fragment order:
// xp[((mt*64+hb)*4+g)*256 + lane*8 + r]  (m = mt*16 + r + 8*(lane>=16),
// h = hb*16 + lane%16).
// ---------------------------------------------------------------------------
__global__ void build_xproj(const float* __restrict__ x,
                            const float* __restrict__ W_ih0,
                            const float* __restrict__ b_ih0,
                            const float* __restrict__ b_hh0,
                            float* __restrict__ xp) {
  int tid  = blockIdx.x * blockDim.x + threadIdx.x;  // 0 .. 2M-1
  int r    = tid & 7;
  int lane = (tid >> 3) & 31;
  int g    = (tid >> 8) & 3;
  int hb   = (tid >> 10) & 63;
  int mt   = tid >> 16;                              // 0..31
  int m    = mt * 16 + r + ((lane >> 4) << 3);
  int h    = hb * 16 + (lane & 15);
  int col  = g * Hd + h;
  xp[tid] = x[m * 2] * W_ih0[col * 2] + x[m * 2 + 1] * W_ih0[col * 2 + 1]
          + b_ih0[col] + b_hh0[col];
}

// ---------------------------------------------------------------------------
// Cooperative A staging: copy this block's 16 h rows (32 KB bf16) to LDS.
// ---------------------------------------------------------------------------
__device__ __forceinline__ void stage_A(__bf16* sA, const __bf16* __restrict__ g,
                                        int mt) {
  const v8bf* src = (const v8bf*)(g + (size_t)mt * 16 * Hd);
  v8bf* dst = (v8bf*)sA;
#pragma unroll
  for (int j = 0; j < 8; ++j)
    dst[threadIdx.x + j * 256] = src[threadIdx.x + j * 256];
}

// A fragment per ISA 16-bit 16x32 layout: lane<16 holds K {0..7,16..23},
// lane>=16 holds K {8..15,24..31} for row M = lane%16.
__device__ __forceinline__ v16bf load_a(const __bf16* arow, int kc) {
  v8bf alo = *(const v8bf*)(arow + kc * 32);
  v8bf ahi = *(const v8bf*)(arow + kc * 32 + 16);
  return __builtin_shufflevector(alo, ahi, 0, 1, 2, 3, 4, 5, 6, 7,
                                 8, 9, 10, 11, 12, 13, 14, 15);
}

// ---------------------------------------------------------------------------
// Depth-4 circular software pipeline over the 32 K-chunks, fenced with hard
// sched_barrier(0): [4x WMMA slot kc%4] || [refill slot with kc+4] || ...
// ---------------------------------------------------------------------------
__device__ __forceinline__ void gemm_acc(v8f acc[4],
                                         const __bf16* sA,
                                         const __bf16* __restrict__ wf,
                                         int hb, int lane) {
  const __bf16* arow  = sA + (lane & 15) * Hd + ((lane >> 4) << 3);
  const __bf16* wbase = wf + (size_t)hb * 4 * 32 * 512 + lane * 16;

  v16bf ab[4];
  v16bf bb[4][4];
#pragma unroll
  for (int i = 0; i < 4; ++i) {
    ab[i] = load_a(arow, i);
#pragma unroll
    for (int g = 0; g < 4; ++g)
      bb[i][g] = *(const v16bf*)(wbase + (size_t)i * 512 + (size_t)g * 32 * 512);
  }
  __builtin_amdgcn_sched_barrier(0);   // keep prologue loads in front

#pragma unroll
  for (int kc = 0; kc < 32; ++kc) {
    const int cur = kc & 3;
#pragma unroll
    for (int g = 0; g < 4; ++g)
      acc[g] = __builtin_amdgcn_wmma_f32_16x16x32_bf16(
          false, ab[cur], false, bb[cur][g], (short)0, acc[g], false, false);
    __builtin_amdgcn_sched_barrier(0); // WMMA group stays before the refill
    if (kc + 4 < 32) {
      ab[cur] = load_a(arow, kc + 4);
      const __bf16* wk = wbase + (size_t)(kc + 4) * 512;
#pragma unroll
      for (int g = 0; g < 4; ++g)
        bb[cur][g] = *(const v16bf*)(wk + (size_t)g * 32 * 512);
      __builtin_amdgcn_sched_barrier(0); // refill stays in this iteration
    }
  }
}

// ---------------------------------------------------------------------------
// Layer 0: gates = xproj + h0 @ W_hh0^T ; fused cell update.
// 2048 waves total: wave wid -> (mt = wid/64, hb = wid%64); the 8 waves of a
// block share mt, so they share the staged A tile.
// ---------------------------------------------------------------------------
__global__ void __launch_bounds__(256) lstm_layer0(
    const __bf16* __restrict__ h0_src, __bf16* __restrict__ h0_dst,
    float* __restrict__ c0, const __bf16* __restrict__ wf,
    const float* __restrict__ xp) {
  __shared__ __bf16 sA[16 * Hd];
  int lane = threadIdx.x & 31;
  int wid  = blockIdx.x * 8 + (threadIdx.x >> 5);
  int mt   = wid >> 6;
  int hb   = wid & 63;

  stage_A(sA, h0_src, mt);

  v8f acc[4];
  const float* xb = xp + ((size_t)(mt * 64 + hb) * 4) * 256 + lane * 8;
#pragma unroll
  for (int g = 0; g < 4; ++g) acc[g] = *(const v8f*)(xb + (size_t)g * 256);

  __syncthreads();
  gemm_acc(acc, sA, wf, hb, lane);

  int h  = hb * 16 + (lane & 15);
  int mb = mt * 16 + ((lane >> 4) << 3);
#pragma unroll
  for (int r = 0; r < 8; ++r) {
    size_t idx = (size_t)(mb + r) * Hd + h;
    float iv = sigf(acc[0][r]);
    float fv = sigf(acc[1][r]);
    float gv = tanhf_(acc[2][r]);
    float ov = sigf(acc[3][r]);
    float c  = fv * c0[idx] + iv * gv;
    c0[idx]  = c;
    h0_dst[idx] = (__bf16)(ov * tanhf_(c));
  }
}

// ---------------------------------------------------------------------------
// Layer 1: gates = h0_new @ W_ih1^T + h1 @ W_hh1^T + b1 ; fused cell update.
// ---------------------------------------------------------------------------
__global__ void __launch_bounds__(256) lstm_layer1(
    const __bf16* __restrict__ h0_new, const __bf16* __restrict__ h1_src,
    __bf16* __restrict__ h1_dst, float* __restrict__ c1,
    float* __restrict__ h1f, const __bf16* __restrict__ wfi,
    const __bf16* __restrict__ wfh, const float* __restrict__ b1p) {
  __shared__ __bf16 sA0[16 * Hd];
  __shared__ __bf16 sA1[16 * Hd];
  int lane = threadIdx.x & 31;
  int wid  = blockIdx.x * 8 + (threadIdx.x >> 5);
  int mt   = wid >> 6;
  int hb   = wid & 63;
  int h    = hb * 16 + (lane & 15);

  stage_A(sA0, h0_new, mt);
  stage_A(sA1, h1_src, mt);

  v8f acc[4];
#pragma unroll
  for (int g = 0; g < 4; ++g) {
    float bv = b1p[g * Hd + h];
#pragma unroll
    for (int r = 0; r < 8; ++r) acc[g][r] = bv;
  }

  __syncthreads();
  gemm_acc(acc, sA0, wfi, hb, lane);
  gemm_acc(acc, sA1, wfh, hb, lane);

  int mb = mt * 16 + ((lane >> 4) << 3);
#pragma unroll
  for (int r = 0; r < 8; ++r) {
    size_t idx = (size_t)(mb + r) * Hd + h;
    float iv = sigf(acc[0][r]);
    float fv = sigf(acc[1][r]);
    float gv = tanhf_(acc[2][r]);
    float ov = sigf(acc[3][r]);
    float c  = fv * c1[idx] + iv * gv;
    c1[idx]  = c;
    float hn = ov * tanhf_(c);
    h1f[idx] = hn;
    h1_dst[idx] = (__bf16)hn;
  }
}

// ---------------------------------------------------------------------------
// FC head: out[m, t, o] = b_fc[o] + sum_h h1[m,h] * W_fc[o,h]   (OUT = 2)
// ---------------------------------------------------------------------------
__global__ void fc_pred(const float* __restrict__ h1f,
                        const float* __restrict__ Wfc,
                        const float* __restrict__ bfc,
                        float* __restrict__ out, int t) {
  int tid = blockIdx.x * blockDim.x + threadIdx.x;  // 0..1023
  int m = tid >> 1, o = tid & 1;
  const v4f* hr = (const v4f*)(h1f + (size_t)m * Hd);
  const v4f* wr = (const v4f*)(Wfc + (size_t)o * Hd);
  float s = bfc[o];
#pragma unroll 4
  for (int k = 0; k < Hd / 4; ++k) {
    v4f a = hr[k], b = wr[k];
    s += a[0] * b[0] + a[1] * b[1] + a[2] * b[2] + a[3] * b[3];
  }
  out[(size_t)m * (T * 2) + t * 2 + o] = s;
}

// ---------------------------------------------------------------------------
extern "C" void kernel_launch(void* const* d_in, const int* in_sizes, int n_in,
                              void* d_out, int out_size, void* d_ws, size_t ws_size,
                              hipStream_t stream) {
  const float* x     = (const float*)d_in[0];
  const float* W_ih0 = (const float*)d_in[1];
  const float* W_hh0 = (const float*)d_in[2];
  const float* b_ih0 = (const float*)d_in[3];
  const float* b_hh0 = (const float*)d_in[4];
  const float* W_ih1 = (const float*)d_in[5];
  const float* W_hh1 = (const float*)d_in[6];
  const float* b_ih1 = (const float*)d_in[7];
  const float* b_hh1 = (const float*)d_in[8];
  const float* W_fc  = (const float*)d_in[9];
  const float* b_fc  = (const float*)d_in[10];
  float* out = (float*)d_out;

  // Workspace carve-up (~44 MB total).
  char* ws = (char*)d_ws;
  size_t off = 0;
  auto carve = [&](size_t bytes) -> void* {
    void* p = ws + off;
    off += (bytes + 255) & ~(size_t)255;
    return p;
  };
  __bf16* wf0   = (__bf16*)carve((size_t)FH * Hd * 2);   // 8 MB
  __bf16* wf1i  = (__bf16*)carve((size_t)FH * Hd * 2);   // 8 MB
  __bf16* wf1h  = (__bf16*)carve((size_t)FH * Hd * 2);   // 8 MB
  float*  xp    = (float*) carve((size_t)Bsz * FH * 4);  // 8 MB
  float*  b1p   = (float*) carve((size_t)FH * 4);
  float*  c0    = (float*) carve((size_t)Bsz * Hd * 4);  // 2 MB
  float*  c1    = (float*) carve((size_t)Bsz * Hd * 4);  // 2 MB
  float*  h1f   = (float*) carve((size_t)Bsz * Hd * 4);  // 2 MB
  __bf16* h0bf[2] = {(__bf16*)carve((size_t)Bsz * Hd * 2),
                     (__bf16*)carve((size_t)Bsz * Hd * 2)};
  __bf16* h1bf[2] = {(__bf16*)carve((size_t)Bsz * Hd * 2),
                     (__bf16*)carve((size_t)Bsz * Hd * 2)};

  // One-time (per call) prep — deterministic, rebuilt every launch.
  init_state<<<(Bsz * Hd) / 256, 256, 0, stream>>>(c0, c1, h0bf[0], h1bf[0],
                                                   b_ih1, b_hh1, b1p);
  swizzle_w<<<(FH * Hd) / 256, 256, 0, stream>>>(W_hh0, wf0);
  swizzle_w<<<(FH * Hd) / 256, 256, 0, stream>>>(W_ih1, wf1i);
  swizzle_w<<<(FH * Hd) / 256, 256, 0, stream>>>(W_hh1, wf1h);
  build_xproj<<<(Bsz * FH) / 256, 256, 0, stream>>>(x, W_ih0, b_ih0, b_hh0, xp);

  // Recurrent chain: 106 steps, ping-pong h buffers; c is owner-updated.
  for (int t = 0; t < T; ++t) {
    int cur = t & 1, nxt = cur ^ 1;
    lstm_layer0<<<256, 256, 0, stream>>>(h0bf[cur], h0bf[nxt], c0, wf0, xp);
    lstm_layer1<<<256, 256, 0, stream>>>(h0bf[nxt], h1bf[cur], h1bf[nxt], c1,
                                         h1f, wf1i, wf1h, b1p);
    fc_pred<<<4, 256, 0, stream>>>(h1f, W_fc, b_fc, out, t);
  }
  (void)in_sizes; (void)n_in; (void)out_size; (void)ws_size;
}
